// ActionModule_71957882077551
// MI455X (gfx1250) — matrix operand order repaired and checked by reading, not compiled, not run on hardware
//
#include <hip/hip_runtime.h>
#include <hip/hip_bf16.h>
#include <math.h>

typedef _Float16 h16;
typedef __attribute__((ext_vector_type(16))) _Float16 v16h;
typedef __attribute__((ext_vector_type(8)))  float    v8f;
typedef __attribute__((__vector_size__(4 * sizeof(int)))) int i32x4;

// CDNA5 async global->LDS path (ASYNCcnt), guarded so either toolchain compiles.
#if __has_builtin(__builtin_amdgcn_global_load_async_to_lds_b128)
#define HAVE_ASYNC_LDS 1
#else
#define HAVE_ASYNC_LDS 0
#endif

#if __has_builtin(__builtin_amdgcn_s_wait_asynccnt)
#define WAIT_ASYNC() __builtin_amdgcn_s_wait_asynccnt(0)
#else
#define WAIT_ASYNC() asm volatile("s_wait_asynccnt 0" ::: "memory")
#endif

namespace {
constexpr int BB   = 2;          // batch
constexpr int TT   = 16;         // tt (frames) == N_feats
constexpr int S    = 1024;       // th*tw
constexpr int L    = TT * S;     // 16384
constexpr int C    = 1536;       // channel
constexpr int CC   = 1024;       // inner dim c
constexpr int HEADS= 16;
constexpr int HD   = 64;         // head dim
constexpr int MROW = BB * S * TT;// 32768 rows through all big GEMMs
constexpr int KPAD = 1568;       // mlp_in 1560 padded to multiple of 32
constexpr int NFRM = 61;         // N_frames
constexpr int PT   = 12;         // pad_t = VAE_T*WIN
constexpr int HID  = 128;        // kb hidden
}

#define DEV __device__ __forceinline__

// branch-free gelu(tanh approx): tanh(u) = 1 - 2/(exp(2u)+1), via v_exp_f32
DEV float gelu_tanh(float x) {
  float u = 0.7978845608028654f * (x + 0.044715f * x * x * x);
  float t = 1.0f - 2.0f / (__expf(2.0f * u) + 1.0f);
  return 0.5f * x * (1.0f + t);
}

// ---------------------------------------------------------------------------
// Weight convert + transpose: src f32 [K,N] -> dst f16 [N,Kpad] (zero-padded K)
// ---------------------------------------------------------------------------
__global__ void convert_transpose_kernel(const float* __restrict__ src,
                                         h16* __restrict__ dst,
                                         int K, int N, int Kpad) {
  long long idx = (long long)blockIdx.x * blockDim.x + threadIdx.x;
  long long total = (long long)N * Kpad;
  if (idx >= total) return;
  int n  = (int)(idx / Kpad);
  int kp = (int)(idx % Kpad);
  float v = (kp < K) ? src[(size_t)kp * N + n] : 0.0f;
  dst[(size_t)n * Kpad + kp] = (h16)v;
}

// ---------------------------------------------------------------------------
// Build gm matrix: f16 [MROW, KPAD]; cols 0..1535 = hidden (x transposed),
// cols 1536..1559 = mouse window features, 1560..1567 = zero pad.
// ---------------------------------------------------------------------------
__global__ void build_gm_kernel(const float* __restrict__ x,
                                const float* __restrict__ mouse,
                                h16* __restrict__ agm) {
  long long gid = (long long)blockIdx.x * blockDim.x + threadIdx.x;
  long long base = gid * 8;
  if (base >= (long long)MROW * KPAD) return;
  int r  = (int)(base / KPAD);
  int c0 = (int)(base % KPAD);           // multiple of 8, never straddles 1536/1560
  int t  = r & (TT - 1);
  int bs = r >> 4;
  int s  = bs & (S - 1);
  int b  = bs >> 10;
  h16 outv[8];
#pragma unroll
  for (int i = 0; i < 8; ++i) {
    int cc = c0 + i;
    float v;
    if (cc < C) {
      v = x[((size_t)b * L + (size_t)t * S + s) * C + cc];
    } else if (cc < C + 2 * PT) {
      int f  = cc - C;
      int j  = f >> 1;
      int cm = f & 1;
      int wi = 4 * t + j;
      int sr = (wi < PT) ? 0 : (wi - PT);
      v = mouse[((size_t)b * NFRM + sr) * 2 + cm];
    } else {
      v = 0.0f;
    }
    outv[i] = (h16)v;
  }
  *reinterpret_cast<float4*>(agm + base) = *reinterpret_cast<float4*>(outv);
}

// ---------------------------------------------------------------------------
// Generic f16 WMMA GEMM:  out = A[M,K] @ Wt[N,K]^T  (Wt stored [N,K])
// Block tile 64x64, 4 waves (128 threads), wave tile 32x32 = 2x2 WMMA tiles,
// K step 32 via v_wmma_f32_16x16x32_f16, f32 accumulation.
// Async path: double-buffered LDS, global_load_async_to_lds_b128 streams the
// next K-tile (ASYNCcnt) while the current one feeds the WMMAs.
// ---------------------------------------------------------------------------
enum { EPI_F32 = 0, EPI_GELU_F16 = 1, EPI_F16 = 2, EPI_ADDX = 3, EPI_ADDOUT = 4 };

template <int EPI>
__global__ __launch_bounds__(128)
void gemm_wmma_f16(const h16* __restrict__ A, const h16* __restrict__ Wt,
                   const float* __restrict__ bias,
                   float* __restrict__ out32, h16* __restrict__ out16,
                   const float* __restrict__ xin,
                   int Mdim, int Ndim, int Kdim) {
  __shared__ h16 As[2][64][32];
  __shared__ h16 Bs[2][64][32];

  const int tid  = threadIdx.x;
  const int lane = tid & 31;
  const int wave = tid >> 5;
  const int wm   = wave >> 1;      // 0..1
  const int wn   = wave & 1;       // 0..1
  const int m0   = blockIdx.y * 64;
  const int n0   = blockIdx.x * 64;
  const int lrow = lane & 15;      // row/col within 16-tile
  const int lhalf= lane >> 4;      // 0/1

  v8f acc[2][2];
#pragma unroll
  for (int mi = 0; mi < 2; ++mi)
#pragma unroll
    for (int ni = 0; ni < 2; ++ni)
#pragma unroll
      for (int j = 0; j < 8; ++j) acc[mi][ni][j] = 0.0f;

  const int arow  = tid >> 1;
  const int acol0 = (tid & 1) * 16;

#if HAVE_ASYNC_LDS
  auto issue_tile = [&](int kk, int buf) {
    const h16* sA = A  + (size_t)(m0 + arow) * Kdim + kk + acol0;
    const h16* sB = Wt + (size_t)(n0 + arow) * Kdim + kk + acol0;
    __builtin_amdgcn_global_load_async_to_lds_b128(
        (i32x4*)sA, (i32x4*)&As[buf][arow][acol0], 0, 0);
    __builtin_amdgcn_global_load_async_to_lds_b128(
        (i32x4*)(sA + 8), (i32x4*)&As[buf][arow][acol0 + 8], 0, 0);
    __builtin_amdgcn_global_load_async_to_lds_b128(
        (i32x4*)sB, (i32x4*)&Bs[buf][arow][acol0], 0, 0);
    __builtin_amdgcn_global_load_async_to_lds_b128(
        (i32x4*)(sB + 8), (i32x4*)&Bs[buf][arow][acol0 + 8], 0, 0);
  };
  issue_tile(0, 0);
  int p = 0;
  for (int k0 = 0; k0 < Kdim; k0 += 32, p ^= 1) {
    WAIT_ASYNC();          // own tile fill done (ASYNCcnt==0)
    __syncthreads();       // all waves' fills published; prev buffer reads done
    if (k0 + 32 < Kdim) issue_tile(k0 + 32, p ^ 1);
#else
  int p = 0;
  for (int k0 = 0; k0 < Kdim; k0 += 32) {
    {
      const h16* sA = A  + (size_t)(m0 + arow) * Kdim + k0 + acol0;
      const h16* sB = Wt + (size_t)(n0 + arow) * Kdim + k0 + acol0;
      const float4* srcA = reinterpret_cast<const float4*>(sA);
      float4* dstA = reinterpret_cast<float4*>(&As[0][arow][acol0]);
      dstA[0] = srcA[0];
      dstA[1] = srcA[1];
      const float4* srcB = reinterpret_cast<const float4*>(sB);
      float4* dstB = reinterpret_cast<float4*>(&Bs[0][arow][acol0]);
      dstB[0] = srcB[0];
      dstB[1] = srcB[1];
      if (k0 + 32 < Kdim) {            // prefetch next K tiles into cache
        __builtin_prefetch(sA + 32, 0, 0);
        __builtin_prefetch(sB + 32, 0, 0);
      }
    }
    __syncthreads();
#endif

    v16h afrag[2], bfrag[2];
#pragma unroll
    for (int mi = 0; mi < 2; ++mi) {
      int rr = wm * 32 + mi * 16 + lrow;
#pragma unroll
      for (int j = 0; j < 4; ++j) {
        afrag[mi][2 * j]         = As[p][rr][lhalf * 8 + 2 * j];
        afrag[mi][2 * j + 1]     = As[p][rr][lhalf * 8 + 2 * j + 1];
        afrag[mi][8 + 2 * j]     = As[p][rr][16 + lhalf * 8 + 2 * j];
        afrag[mi][8 + 2 * j + 1] = As[p][rr][16 + lhalf * 8 + 2 * j + 1];
      }
    }
#pragma unroll
    for (int ni = 0; ni < 2; ++ni) {
      int rr = wn * 32 + ni * 16 + lrow;
#pragma unroll
      for (int j = 0; j < 8; ++j) {
        bfrag[ni][2 * j]     = Bs[p][rr][(j >> 2) * 8 + lhalf * 8 + 2 * j];
        bfrag[ni][2 * j + 1] = Bs[p][rr][(j >> 2) * 8 + lhalf * 8 + 2 * j + 1];
      }
    }
#pragma unroll
    for (int mi = 0; mi < 2; ++mi)
#pragma unroll
      for (int ni = 0; ni < 2; ++ni)
        acc[mi][ni] = __builtin_amdgcn_wmma_f32_16x16x32_f16(
            false, afrag[mi], false, bfrag[ni], (short)0, acc[mi][ni],
            false, false);
#if !HAVE_ASYNC_LDS
    __syncthreads();
#endif
  }

  // epilogue
#pragma unroll
  for (int mi = 0; mi < 2; ++mi) {
#pragma unroll
    for (int ni = 0; ni < 2; ++ni) {
      int gc = n0 + wn * 32 + ni * 16 + lrow;
#pragma unroll
      for (int j = 0; j < 8; ++j) {
        int gr = m0 + wm * 32 + mi * 16 + lhalf * 8 + j;
        float v = acc[mi][ni][j];
        if (EPI == EPI_F32) {
          if (bias) v += bias[gc];
          out32[(size_t)gr * Ndim + gc] = v;
        } else if (EPI == EPI_GELU_F16) {
          v = gelu_tanh(v + bias[gc]);
          out16[(size_t)gr * Ndim + gc] = (h16)v;
        } else if (EPI == EPI_F16) {
          out16[(size_t)gr * Ndim + gc] = (h16)v;
        } else if (EPI == EPI_ADDX) {
          // gr = (b*1024+s)*16 + t  ->  xrow = b*L + t*S + s
          int t  = gr & 15;
          int bs = gr >> 4;
          int s  = bs & (S - 1);
          int b  = bs >> 10;
          size_t xrow = (size_t)b * L + (size_t)t * S + s;
          float o = xin[xrow * C + gc] + v;
          out32[xrow * C + gc] = o;          // d_out = x + mouse_out
          out16[xrow * C + gc] = (h16)o;     // f16 copy for next GEMM
        } else { // EPI_ADDOUT
          out32[(size_t)gr * Ndim + gc] += v;
        }
      }
    }
  }
}

// ---------------------------------------------------------------------------
// LayerNorm over 1024: f32 in -> f16 out
// ---------------------------------------------------------------------------
__global__ __launch_bounds__(256)
void layernorm_kernel(const float* __restrict__ t2, const float* __restrict__ g,
                      const float* __restrict__ bta, h16* __restrict__ hln) {
  int row = blockIdx.x;
  size_t base = (size_t)row * CC;
  int tid = threadIdx.x;
  float s1 = 0.f, s2 = 0.f;
  for (int i = tid; i < CC; i += 256) {
    float v = t2[base + i];
    s1 += v; s2 += v * v;
  }
  for (int m = 16; m; m >>= 1) {
    s1 += __shfl_xor(s1, m, 32);
    s2 += __shfl_xor(s2, m, 32);
  }
  __shared__ float r1[8], r2[8];
  int wid = tid >> 5, lane = tid & 31;
  if (lane == 0) { r1[wid] = s1; r2[wid] = s2; }
  __syncthreads();
  if (tid == 0) {
    float a = 0.f, b = 0.f;
    for (int w = 0; w < 8; ++w) { a += r1[w]; b += r2[w]; }
    r1[0] = a; r2[0] = b;
  }
  __syncthreads();
  float mu  = r1[0] * (1.0f / CC);
  float var = r2[0] * (1.0f / CC) - mu * mu;
  float rstd = rsqrtf(var + 1e-5f);
  for (int i = tid; i < CC; i += 256) {
    float v = (t2[base + i] - mu) * rstd * g[i] + bta[i];
    hln[base + i] = (h16)v;
  }
}

// ---------------------------------------------------------------------------
// Attention 1 (mouse): per (bs, head). qkv f16 [MROW, 3072]; rms + rope +
// local-causal(6) softmax over 16 frames; out f16 [MROW, 1024].
// ---------------------------------------------------------------------------
__global__ __launch_bounds__(256)
void attn1_kernel(const h16* __restrict__ qkvh, h16* __restrict__ att1) {
  __shared__ float qs[16][64], ks[16][64], vs[16][64], ps[16][16];
  const int head = blockIdx.x;
  const int bs   = blockIdx.y;
  const int tid  = threadIdx.x;
  const int t  = tid >> 4;
  const int g  = tid & 15;
  const int d0 = g * 4;

  const h16* qp = qkvh + (size_t)(bs * TT + t) * 3072 + head * HD + d0;
#pragma unroll
  for (int i = 0; i < 4; ++i) {
    qs[t][d0 + i] = (float)qp[i];
    ks[t][d0 + i] = (float)qp[1024 + i];
    vs[t][d0 + i] = (float)qp[2048 + i];
  }
  __syncthreads();

  // rms-norm q and k rows (dim 64), groups of 16 lanes share row t
  float ssq = 0.f, ssk = 0.f;
#pragma unroll
  for (int i = 0; i < 4; ++i) {
    ssq += qs[t][d0 + i] * qs[t][d0 + i];
    ssk += ks[t][d0 + i] * ks[t][d0 + i];
  }
  for (int m = 8; m; m >>= 1) {
    ssq += __shfl_xor(ssq, m, 16);
    ssk += __shfl_xor(ssk, m, 16);
  }
  float rq = rsqrtf(ssq * (1.0f / HD) + 1e-6f);
  float rk = rsqrtf(ssk * (1.0f / HD) + 1e-6f);
#pragma unroll
  for (int i = 0; i < 4; ++i) { qs[t][d0 + i] *= rq; ks[t][d0 + i] *= rk; }

  // rope: this thread owns pairs p = d0/2, d0/2+1
#pragma unroll
  for (int pp = 0; pp < 2; ++pp) {
    int p = d0 / 2 + pp;
    float e;
    if (p < 4)       e = p * 0.25f;
    else if (p < 18) e = (p - 4) * (1.0f / 14.0f);
    else             e = (p - 18) * (1.0f / 14.0f);
    float f   = __expf(-e * 5.545177444479562f);   // 256^-e
    float ang = (float)t * f;
    float cs = __cosf(ang), sn = __sinf(ang);
    float qr = qs[t][2 * p], qi = qs[t][2 * p + 1];
    qs[t][2 * p]     = qr * cs - qi * sn;
    qs[t][2 * p + 1] = qi * cs + qr * sn;
    float kr = ks[t][2 * p], ki = ks[t][2 * p + 1];
    ks[t][2 * p]     = kr * cs - ki * sn;
    ks[t][2 * p + 1] = ki * cs + kr * sn;
  }
  __syncthreads();

  // scores + softmax: thread (t, j)
  {
    int j = g;
    float s = 0.f;
#pragma unroll
    for (int d = 0; d < HD; ++d) s += qs[t][d] * ks[j][d];
    s *= 0.125f;
    bool valid = (j <= t) && (t - j < 6);
    s = valid ? s : -1e30f;
    float mx = s;
    for (int m = 8; m; m >>= 1) mx = fmaxf(mx, __shfl_xor(mx, m, 16));
    float e = __expf(s - mx);
    float sm = e;
    for (int m = 8; m; m >>= 1) sm += __shfl_xor(sm, m, 16);
    ps[t][j] = e / sm;
  }
  __syncthreads();

  float o[4] = {0.f, 0.f, 0.f, 0.f};
  for (int j = 0; j < 16; ++j) {
    float pj = ps[t][j];
#pragma unroll
    for (int i = 0; i < 4; ++i) o[i] += pj * vs[j][d0 + i];
  }
  h16* op = att1 + (size_t)(bs * TT + t) * CC + head * HD + d0;
#pragma unroll
  for (int i = 0; i < 4; ++i) op[i] = (h16)o[i];
}

// ---------------------------------------------------------------------------
// Keyboard MLP: kb = silu(kbc @ w1 + b1) @ w2 + b2  -> kbbuf [B*61, 128]
// ---------------------------------------------------------------------------
__global__ __launch_bounds__(128)
void kb_mlp_kernel(const float* __restrict__ kbc, const float* __restrict__ w1,
                   const float* __restrict__ b1, const float* __restrict__ w2,
                   const float* __restrict__ b2, float* __restrict__ kbbuf) {
  __shared__ float h1s[HID];
  int row = blockIdx.x;       // 0..121
  int tid = threadIdx.x;
  float acc = b1[tid];
  for (int i = 0; i < 6; ++i) acc += kbc[(size_t)row * 6 + i] * w1[i * HID + tid];
  h1s[tid] = acc / (1.0f + __expf(-acc));
  __syncthreads();
  float o = b2[tid];
  for (int i = 0; i < HID; ++i) o += h1s[i] * w2[i * HID + tid];
  kbbuf[(size_t)row * HID + tid] = o;
}

// ---------------------------------------------------------------------------
// Build gk row (per b,t) + kv = gk @ kb_kv_w  -> kvbuf f32 [B,16,2,16,64]
// ---------------------------------------------------------------------------
__global__ __launch_bounds__(256)
void kb_kv_kernel(const float* __restrict__ kbbuf, const float* __restrict__ kvw,
                  float* __restrict__ kv) {
  __shared__ float gks[1536];
  int b = blockIdx.x >> 4;
  int t = blockIdx.x & 15;
  int tid = threadIdx.x;
  for (int idx = tid; idx < 1536; idx += 256) {
    int j = idx / HID, u = idx % HID;
    int wi = 4 * t + j;
    int sr = (wi < PT) ? 0 : (wi - PT);
    gks[idx] = kbbuf[(size_t)(b * NFRM + sr) * HID + u];
  }
  __syncthreads();
  for (int nn = 0; nn < 8; ++nn) {
    int n = tid + 256 * nn;
    float acc = 0.f;
    for (int k = 0; k < 1536; ++k) acc += gks[k] * kvw[(size_t)k * 2048 + n];
    kv[(size_t)(b * TT + t) * 2048 + n] = acc;
  }
}

// ---------------------------------------------------------------------------
// RMS-norm kk in place: rows (b, j, head) of 64
// ---------------------------------------------------------------------------
__global__ __launch_bounds__(64)
void kk_rms_kernel(float* __restrict__ kv) {
  int rr = blockIdx.x;            // 0..511
  int b = rr >> 8, j = (rr >> 4) & 15, h = rr & 15;
  size_t base = ((size_t)(b * TT + j) * 2) * CC + h * HD;
  int tid = threadIdx.x;
  float v = kv[base + tid];
  float s = v * v;
  for (int m = 16; m; m >>= 1) s += __shfl_xor(s, m, 32);
  __shared__ float ws2[2];
  int wid = tid >> 5, lane = tid & 31;
  if (lane == 0) ws2[wid] = s;
  __syncthreads();
  float tot = ws2[0] + ws2[1];
  kv[base + tid] = v * rsqrtf(tot * (1.0f / HD) + 1e-6f);
}

// ---------------------------------------------------------------------------
// Attention 2 (keyboard): per (b, l). q rms from qk f32; kk pre-normalized.
// ---------------------------------------------------------------------------
__global__ __launch_bounds__(256)
void attn2_kernel(const float* __restrict__ qk, const float* __restrict__ kv,
                  h16* __restrict__ att2) {
  __shared__ float qs2[16][64], ps2[16][16];
  int b = blockIdx.x >> 14;
  int l = blockIdx.x & (L - 1);
  int qf = l >> 10;               // l / S
  int tid = threadIdx.x;
  int h  = tid >> 4;
  int g  = tid & 15;
  int d0 = g * 4;

  const float* qp = qk + (size_t)(b * L + l) * CC + h * HD + d0;
  float v0 = qp[0], v1 = qp[1], v2 = qp[2], v3 = qp[3];
  float ss = v0 * v0 + v1 * v1 + v2 * v2 + v3 * v3;
  for (int m = 8; m; m >>= 1) ss += __shfl_xor(ss, m, 16);
  float rinv = rsqrtf(ss * (1.0f / HD) + 1e-6f);
  qs2[h][d0 + 0] = v0 * rinv;
  qs2[h][d0 + 1] = v1 * rinv;
  qs2[h][d0 + 2] = v2 * rinv;
  qs2[h][d0 + 3] = v3 * rinv;
  __syncthreads();

  {
    int j = g;
    const float* kkb = kv + ((size_t)(b * TT + j) * 2) * CC + h * HD;
    float s = 0.f;
#pragma unroll
    for (int d = 0; d < HD; ++d) s += qs2[h][d] * kkb[d];
    s *= 0.125f;
    bool valid = (j <= qf) && (qf - j < 6);
    s = valid ? s : -1e30f;
    float mx = s;
    for (int m = 8; m; m >>= 1) mx = fmaxf(mx, __shfl_xor(mx, m, 16));
    float e = __expf(s - mx);
    float sm = e;
    for (int m = 8; m; m >>= 1) sm += __shfl_xor(sm, m, 16);
    ps2[h][j] = e / sm;
  }
  __syncthreads();

  float o[4] = {0.f, 0.f, 0.f, 0.f};
  for (int j = 0; j < 16; ++j) {
    float pj = ps2[h][j];
    const float* vvb = kv + ((size_t)(b * TT + j) * 2 + 1) * CC + h * HD + d0;
#pragma unroll
    for (int i = 0; i < 4; ++i) o[i] += pj * vvb[i];
  }
  h16* op = att2 + (size_t)(b * L + l) * CC + h * HD + d0;
#pragma unroll
  for (int i = 0; i < 4; ++i) op[i] = (h16)o[i];
}

// ---------------------------------------------------------------------------
extern "C" void kernel_launch(void* const* d_in, const int* in_sizes, int n_in,
                              void* d_out, int out_size, void* d_ws, size_t ws_size,
                              hipStream_t stream) {
  const float* x      = (const float*)d_in[0];
  const float* mouse  = (const float*)d_in[4];
  const float* kbc    = (const float*)d_in[5];
  const float* kb_w1  = (const float*)d_in[6];
  const float* kb_b1  = (const float*)d_in[7];
  const float* kb_w2  = (const float*)d_in[8];
  const float* kb_b2  = (const float*)d_in[9];
  const float* mm_w1  = (const float*)d_in[10];
  const float* mm_b1  = (const float*)d_in[11];
  const float* mm_w2  = (const float*)d_in[12];
  const float* mm_b2  = (const float*)d_in[13];
  const float* ln_g   = (const float*)d_in[14];
  const float* ln_b   = (const float*)d_in[15];
  const float* tqkv_w = (const float*)d_in[16];
  const float* projm_w= (const float*)d_in[17];
  const float* mq_w   = (const float*)d_in[18];
  const float* kbkv_w = (const float*)d_in[19];
  const float* projk_w= (const float*)d_in[20];
  float* out = (float*)d_out;

  char* ws = (char*)d_ws;
  size_t off = 0;
  auto alloc = [&](size_t bytes) {
    size_t r = off;
    off = (off + bytes + 255) & ~(size_t)255;
    return r;
  };
  size_t O_AGM  = alloc((size_t)MROW * KPAD * 2);   // also reused as x1h f16
  size_t O_W1H  = alloc((size_t)KPAD * CC * 2);
  size_t O_W2H  = alloc((size_t)CC * CC * 2);
  size_t O_WQKV = alloc((size_t)CC * 3072 * 2);
  size_t O_WPM  = alloc((size_t)C * CC * 2);        // proj_mouse^T [1536,1024]
  size_t O_WMQ  = alloc((size_t)CC * C * 2);        // mouse_q^T    [1024,1536]
  size_t O_WPK  = alloc((size_t)C * CC * 2);        // proj_kb^T    [1536,1024]
  size_t O_H1   = alloc((size_t)MROW * CC * 2);     // also reused as attn1
  size_t O_T2   = alloc((size_t)MROW * CC * 4);     // also reused as qk
  size_t O_HLN  = alloc((size_t)MROW * CC * 2);     // also reused as attn2
  size_t O_QKV  = alloc((size_t)MROW * 3072 * 2);
  size_t O_KB   = alloc((size_t)BB * NFRM * HID * 4);
  size_t O_KV   = alloc((size_t)BB * TT * 2048 * 4);
  (void)ws_size;

  h16*   AGM  = (h16*)(ws + O_AGM);
  h16*   X1H  = (h16*)(ws + O_AGM);
  h16*   W1H  = (h16*)(ws + O_W1H);
  h16*   W2H  = (h16*)(ws + O_W2H);
  h16*   WQKV = (h16*)(ws + O_WQKV);
  h16*   WPM  = (h16*)(ws + O_WPM);
  h16*   WMQ  = (h16*)(ws + O_WMQ);
  h16*   WPK  = (h16*)(ws + O_WPK);
  h16*   H1   = (h16*)(ws + O_H1);
  h16*   ATT1 = (h16*)(ws + O_H1);
  float* T2   = (float*)(ws + O_T2);
  float* QK   = (float*)(ws + O_T2);
  h16*   HLN  = (h16*)(ws + O_HLN);
  h16*   ATT2 = (h16*)(ws + O_HLN);
  h16*   QKVH = (h16*)(ws + O_QKV);
  float* KBUF = (float*)(ws + O_KB);
  float* KV   = (float*)(ws + O_KV);

  auto cgrid = [](long long n) { return (unsigned)((n + 255) / 256); };

  // weight convert+transpose (f32 [K,N] -> f16 [N,Kpad])
  convert_transpose_kernel<<<cgrid((long long)CC * KPAD), 256, 0, stream>>>(mm_w1,  W1H,  1560, CC,   KPAD);
  convert_transpose_kernel<<<cgrid((long long)CC * CC),   256, 0, stream>>>(mm_w2,  W2H,  CC,   CC,   CC);
  convert_transpose_kernel<<<cgrid((long long)3072 * CC), 256, 0, stream>>>(tqkv_w, WQKV, CC,   3072, CC);
  convert_transpose_kernel<<<cgrid((long long)C * CC),    256, 0, stream>>>(projm_w,WPM,  CC,   C,    CC);
  convert_transpose_kernel<<<cgrid((long long)CC * C),    256, 0, stream>>>(mq_w,   WMQ,  C,    CC,   C);
  convert_transpose_kernel<<<cgrid((long long)C * CC),    256, 0, stream>>>(projk_w,WPK,  CC,   C,    CC);

  // build gm input matrix
  build_gm_kernel<<<cgrid((long long)MROW * KPAD / 8), 256, 0, stream>>>(x, mouse, AGM);

  // G1: gelu(gm @ mm_w1 + b1) -> H1 f16
  gemm_wmma_f16<EPI_GELU_F16><<<dim3(CC / 64, MROW / 64), 128, 0, stream>>>(
      AGM, W1H, mm_b1, nullptr, H1, nullptr, MROW, CC, KPAD);
  // G2: H1 @ mm_w2 + b2 -> T2 f32
  gemm_wmma_f16<EPI_F32><<<dim3(CC / 64, MROW / 64), 128, 0, stream>>>(
      H1, W2H, mm_b2, T2, nullptr, nullptr, MROW, CC, CC);
  // LayerNorm -> HLN f16
  layernorm_kernel<<<MROW, 256, 0, stream>>>(T2, ln_g, ln_b, HLN);
  // G3: HLN @ t_qkv_w -> QKVH f16
  gemm_wmma_f16<EPI_F16><<<dim3(3072 / 64, MROW / 64), 128, 0, stream>>>(
      HLN, WQKV, nullptr, nullptr, QKVH, nullptr, MROW, 3072, CC);
  // A1: mouse attention -> ATT1 (reuses H1)
  attn1_kernel<<<dim3(HEADS, BB * S), 256, 0, stream>>>(QKVH, ATT1);
  // G4: ATT1 @ proj_mouse, scatter-add x -> d_out f32 + X1H f16 (reuses AGM)
  gemm_wmma_f16<EPI_ADDX><<<dim3(C / 64, MROW / 64), 128, 0, stream>>>(
      ATT1, WPM, nullptr, out, X1H, x, MROW, C, CC);

  // keyboard branch
  kb_mlp_kernel<<<BB * NFRM, HID, 0, stream>>>(kbc, kb_w1, kb_b1, kb_w2, kb_b2, KBUF);
  kb_kv_kernel<<<BB * TT, 256, 0, stream>>>(KBUF, kbkv_w, KV);
  kk_rms_kernel<<<BB * TT * HEADS, 64, 0, stream>>>(KV);

  // G5: X1H @ mouse_q_w -> QK f32 (reuses T2)
  gemm_wmma_f16<EPI_F32><<<dim3(CC / 64, MROW / 64), 128, 0, stream>>>(
      X1H, WMQ, nullptr, QK, nullptr, nullptr, MROW, CC, C);
  // A2: keyboard attention -> ATT2 (reuses HLN)
  attn2_kernel<<<BB * L, 256, 0, stream>>>(QK, KV, ATT2);
  // G6: ATT2 @ proj_kb, add in place -> d_out
  gemm_wmma_f16<EPI_ADDOUT><<<dim3(C / 64, MROW / 64), 128, 0, stream>>>(
      ATT2, WPK, nullptr, out, nullptr, nullptr, MROW, C, CC);
}
